// MultiHeadAttention_29944511987983
// MI455X (gfx1250) — matrix operand order repaired and checked by reading, not compiled, not run on hardware
//
#include <hip/hip_runtime.h>
#include <hip/hip_bf16.h>

// MI455X (gfx1250) multi-head attention forward, bf16 WMMA + f32 accumulate.
//
// Workspace layout (needs ~88 MiB of d_ws):
//   xb [8192,1024] bf16 | wqb [1024,3072] bf16 | wob [1024,1024] bf16
//   Q/K/V [4,16,2048,64] bf16 | ctx [8192,1024] bf16 ([B,T,H,d] == [B*T,C])

typedef __bf16 bf16_t;
typedef __attribute__((ext_vector_type(16))) __bf16 v16bf;
typedef __attribute__((ext_vector_type(8)))  __bf16 v8bf;
typedef __attribute__((ext_vector_type(8)))  float  v8f;
typedef __attribute__((ext_vector_type(4)))  int    v4i_t;

#define D_MODEL 1024
#define N_HEADS 16
#define D_HEAD  64
#define BATCH   4
#define SEQ     2048
#define M_TOTAL (BATCH * SEQ)     // 8192
#define N_QKV   (3 * D_MODEL)     // 3072

// CDNA5 async global->LDS (ASYNCcnt) path, guarded so compile never breaks.
#if __has_builtin(__builtin_amdgcn_global_load_async_to_lds_b128) && \
    __has_builtin(__builtin_amdgcn_s_wait_asynccnt)
#define USE_ASYNC_LDS 1
#endif

__device__ inline void copy16_g2lds(void* lds, const void* g) {
#if defined(USE_ASYNC_LDS)
  // Signature (from hipcc diagnostic): arg0 = v4i addrspace(1)* (global src),
  // arg1 = v4i addrspace(3)* (LDS dst), imm offset, imm cpol.
  __builtin_amdgcn_global_load_async_to_lds_b128(
      (__attribute__((address_space(1))) v4i_t*)g,
      (__attribute__((address_space(3))) v4i_t*)lds, /*offset=*/0, /*cpol=*/0);
#else
  *(uint4*)lds = *(const uint4*)g;
#endif
}

__device__ inline void wait_async_copies() {
#if defined(USE_ASYNC_LDS)
  __builtin_amdgcn_s_wait_asynccnt(0);
#endif
}

// ---------------------------------------------------------------------------
// f32 -> bf16 cast (grid-stride)
// ---------------------------------------------------------------------------
__global__ __launch_bounds__(256) void cast_f32_bf16(
    const float* __restrict__ src, bf16_t* __restrict__ dst, int n) {
  int i = blockIdx.x * blockDim.x + threadIdx.x;
  int stride = gridDim.x * blockDim.x;
  for (; i < n; i += stride) dst[i] = (bf16_t)src[i];
}

// ---------------------------------------------------------------------------
// WMMA helpers (CDNA5 ISA 7.12.2 VGPR layouts, wave32)
// ---------------------------------------------------------------------------
__device__ inline v8f wmma_bf16(v16bf a, v16bf b, v8f c) {
  return __builtin_amdgcn_wmma_f32_16x16x32_bf16(
      false, a, false, b, (short)0, c, false, false);
}

// A 16x32 bf16 from row-major LDS [M][ld]: lane L -> row L%16;
// lanes 0-15 carry K=0..15; lanes 16-31 carry K=16..31.
__device__ inline v16bf load_a_frag(const bf16_t* __restrict__ base, int ld,
                                    int row0, int lane) {
  const bf16_t* r = base + (row0 + (lane & 15)) * ld;
  int kg = (lane >> 4) * 16;
  v16bf a;
#pragma unroll
  for (int e = 0; e < 8; ++e) a[e] = r[kg + e];          // 16B run
#pragma unroll
  for (int e = 0; e < 8; ++e) a[8 + e] = r[kg + 8 + e]; // 16B run
  return a;
}

// B 32x16 bf16 from FRAGMENT-MAJOR (transposed) LDS [N][ld]: lane L -> col
// N=L%16; lanes 0-15 carry K=0..15, lanes 16-31 carry K=16..31.
// Entire fragment is one contiguous 32B run -> 2x ds_load_b128.
__device__ inline v16bf load_b_frag_tr(const bf16_t* __restrict__ base, int ld,
                                       int col0, int lane) {
  const bf16_t* r = base + (col0 + (lane & 15)) * ld + (lane >> 4) * 16;
  v16bf b;
#pragma unroll
  for (int e = 0; e < 16; ++e) b[e] = r[e];
  return b;
}

// ---------------------------------------------------------------------------
// Shared GEMM mainloop: C[64x128] block tile, 8 waves each owning 32x32.
// A [M,1024] bf16 row-major, B [1024,NSTRIDE] bf16 row-major.
// A tile staged natural (async b128); B tile staged transposed [128][32+pad].
// ---------------------------------------------------------------------------
template <int NSTRIDE>
__device__ inline void gemm_mainloop(const bf16_t* __restrict__ X,
                                     const bf16_t* __restrict__ W,
                                     int bm, int bn, int tid,
                                     bf16_t (*As)[40], bf16_t (*Bst)[40],
                                     v8f acc[2][2]) {
  int lane = tid & 31;
  int wave = tid >> 5;
  int wm = wave >> 2;  // 0..1 row macro-tile
  int wn = wave & 3;   // 0..3 col macro-tile
  int ar = tid >> 2, ac = (tid & 3) * 8;   // A: 64 x 32, 16B per thread
  int br = tid >> 3, bc = (tid & 7) * 16;  // B: 32 x 128, 32B per thread

  for (int k0 = 0; k0 < D_MODEL; k0 += 32) {
    if (k0 + 32 < D_MODEL)  // prefetch next W tile (global_prefetch_b8)
      __builtin_prefetch(&W[(size_t)(k0 + 32 + br) * NSTRIDE + bn + bc], 0, 1);

    // A tile: natural layout, async DMA to LDS when available.
    copy16_g2lds(&As[ar][ac], &X[(size_t)(bm + ar) * D_MODEL + k0 + ac]);
    // B tile: read 32B row chunk, scatter transposed into LDS.
    v16bf wrow = *(const v16bf*)&W[(size_t)(k0 + br) * NSTRIDE + bn + bc];
#pragma unroll
    for (int e = 0; e < 16; ++e) Bst[bc + e][br] = wrow[e];

    wait_async_copies();
    __syncthreads();

    v16bf a0 = load_a_frag(&As[0][0], 40, wm * 32, lane);
    v16bf a1 = load_a_frag(&As[0][0], 40, wm * 32 + 16, lane);
    v16bf b0 = load_b_frag_tr(&Bst[0][0], 40, wn * 32, lane);
    v16bf b1 = load_b_frag_tr(&Bst[0][0], 40, wn * 32 + 16, lane);
    acc[0][0] = wmma_bf16(a0, b0, acc[0][0]);
    acc[0][1] = wmma_bf16(a0, b1, acc[0][1]);
    acc[1][0] = wmma_bf16(a1, b0, acc[1][0]);
    acc[1][1] = wmma_bf16(a1, b1, acc[1][1]);
    __syncthreads();
  }
}

// ---------------------------------------------------------------------------
// QKV projection: [8192,1024] x [1024,3072] -> scatter into Q/K/V [B,H,T,64]
// ---------------------------------------------------------------------------
__global__ __launch_bounds__(256) void qkv_gemm(
    const bf16_t* __restrict__ X, const bf16_t* __restrict__ W,
    bf16_t* __restrict__ Q, bf16_t* __restrict__ K, bf16_t* __restrict__ V) {
  __shared__ alignas(16) bf16_t As[64][40];
  __shared__ alignas(16) bf16_t Bst[128][40];
  int tid = threadIdx.x, lane = tid & 31, wave = tid >> 5;
  int wm = wave >> 2, wn = wave & 3;
  int bm = blockIdx.y * 64, bn = blockIdx.x * 128;
  v8f acc[2][2] = {};
  gemm_mainloop<N_QKV>(X, W, bm, bn, tid, As, Bst, acc);

  int lrow = (lane >> 4) * 8;  // C layout: lanes 16-31 hold rows +8
  int lcol = lane & 15;
#pragma unroll
  for (int i = 0; i < 2; ++i)
#pragma unroll
    for (int j = 0; j < 2; ++j)
#pragma unroll
      for (int r = 0; r < 8; ++r) {
        int m = bm + wm * 32 + i * 16 + lrow + r;
        int n = bn + wn * 32 + j * 16 + lcol;
        int region = n >> 10;  // 0:Q 1:K 2:V
        int c = n & 1023;
        int h = c >> 6, d = c & 63;
        int b = m >> 11, t = m & 2047;
        bf16_t* dst = (region == 0) ? Q : ((region == 1) ? K : V);
        dst[((((size_t)b * N_HEADS + h) * SEQ) + t) * D_HEAD + d] =
            (bf16_t)acc[i][j][r];
      }
}

// ---------------------------------------------------------------------------
// Flash attention: one wave = 16 query rows; block = 8 waves = 128 rows.
// K tile staged natural [32][64+pad] (frag reads contiguous along head-dim),
// V tile staged transposed [64][32+pad] (frag reads contiguous along keys).
// ---------------------------------------------------------------------------
__global__ __launch_bounds__(256) void attn_kernel(
    const bf16_t* __restrict__ Q, const bf16_t* __restrict__ K,
    const bf16_t* __restrict__ V, bf16_t* __restrict__ ctx) {
  __shared__ alignas(16) bf16_t Ks[32][72];
  __shared__ alignas(16) bf16_t Vt[64][40];
  __shared__ alignas(16) bf16_t Pbuf[8][16][32];
  int tid = threadIdx.x, lane = tid & 31, wave = tid >> 5;
  int bh = blockIdx.y;
  int b = bh >> 4, h = bh & 15;
  int qt0 = blockIdx.x * 128 + wave * 16;
  const bf16_t* Qh = Q + (size_t)bh * SEQ * D_HEAD;
  const bf16_t* Kh = K + (size_t)bh * SEQ * D_HEAD;
  const bf16_t* Vh = V + (size_t)bh * SEQ * D_HEAD;

  // Q tile 16x64 as two A-fragments (head-dim 0..31 / 32..63).
  int am = qt0 + (lane & 15);
  int akg = (lane >> 4) * 16;
  v16bf aq[2];
#pragma unroll
  for (int s = 0; s < 2; ++s) {
    const bf16_t* qr = Qh + (size_t)am * D_HEAD + s * 32;
#pragma unroll
    for (int e = 0; e < 8; ++e) aq[s][e] = qr[akg + e];
#pragma unroll
    for (int e = 0; e < 8; ++e) aq[s][8 + e] = qr[akg + 8 + e];
  }

  v8f o[4] = {};
  float mrun[8], lrun[8];
#pragma unroll
  for (int r = 0; r < 8; ++r) { mrun[r] = -1e30f; lrun[r] = 0.f; }

  int bn_ = lane & 15;
  int bkg = (lane >> 4) * 16;
  const float scale = 0.125f;  // 1/sqrt(64)

  // staging indices: 2048 elems per tile, 8 elems (16B) per thread
  int kr = tid >> 3, kc = (tid & 7) * 8;

  for (int j = 0; j < SEQ; j += 32) {
    // ---- cooperative stage of K (async, natural) and V (transposed) ----
    copy16_g2lds(&Ks[kr][kc], &Kh[(size_t)(j + kr) * D_HEAD + kc]);
    v8bf vrow = *(const v8bf*)&Vh[(size_t)(j + kr) * D_HEAD + kc];
#pragma unroll
    for (int e = 0; e < 8; ++e) Vt[kc + e][kr] = vrow[e];
    wait_async_copies();
    __syncthreads();

    // ---- S = Q K^T : 16x32 score tile (two 16x16 col tiles, 4 WMMAs) ----
    v8f s0 = {}, s1 = {};
#pragma unroll
    for (int s = 0; s < 2; ++s) {
      v16bf bk0, bk1;
      const bf16_t* k0p = &Ks[bn_][s * 32 + bkg];
      const bf16_t* k1p = &Ks[16 + bn_][s * 32 + bkg];
#pragma unroll
      for (int e = 0; e < 16; ++e) { bk0[e] = k0p[e]; bk1[e] = k1p[e]; }
      s0 = wmma_bf16(aq[s], bk0, s0);
      s1 = wmma_bf16(aq[s], bk1, s1);
    }

    // ---- online softmax (each row lives inside one 16-lane half) ----
    float mloc[8];
#pragma unroll
    for (int r = 0; r < 8; ++r) {
      s0[r] *= scale; s1[r] *= scale;
      mloc[r] = fmaxf(s0[r], s1[r]);
    }
#pragma unroll
    for (int off = 1; off < 16; off <<= 1)
#pragma unroll
      for (int r = 0; r < 8; ++r)
        mloc[r] = fmaxf(mloc[r], __shfl_xor(mloc[r], off, 32));

    float alpha[8], rs[8];
#pragma unroll
    for (int r = 0; r < 8; ++r) {
      float mnew = fmaxf(mrun[r], mloc[r]);
      alpha[r] = __expf(mrun[r] - mnew);
      mrun[r] = mnew;
      s0[r] = __expf(s0[r] - mnew);
      s1[r] = __expf(s1[r] - mnew);
      rs[r] = s0[r] + s1[r];
    }
#pragma unroll
    for (int off = 1; off < 16; off <<= 1)
#pragma unroll
      for (int r = 0; r < 8; ++r)
        rs[r] += __shfl_xor(rs[r], off, 32);
#pragma unroll
    for (int r = 0; r < 8; ++r) {
      lrun[r] = lrun[r] * alpha[r] + rs[r];
      o[0][r] *= alpha[r]; o[1][r] *= alpha[r];
      o[2][r] *= alpha[r]; o[3][r] *= alpha[r];
    }

    // ---- P: C-layout -> A-layout via per-wave LDS bounce ----
    {
      int prow = (lane >> 4) * 8, pcol = lane & 15;
#pragma unroll
      for (int r = 0; r < 8; ++r) {
        Pbuf[wave][prow + r][pcol]      = (bf16_t)s0[r];
        Pbuf[wave][prow + r][pcol + 16] = (bf16_t)s1[r];
      }
    }
    asm volatile("s_wait_dscnt 0" ::: "memory");  // DS RAW within this wave
    v16bf pa;
    {
      const bf16_t* pr = &Pbuf[wave][lane & 15][0];
#pragma unroll
      for (int e = 0; e < 8; ++e) pa[e] = pr[akg + e];
#pragma unroll
      for (int e = 0; e < 8; ++e) pa[8 + e] = pr[akg + 8 + e];
    }

    // ---- O += P V over 4 head-dim tiles (frags contiguous in Vt) ----
#pragma unroll
    for (int t = 0; t < 4; ++t) {
      v16bf bv;
      const bf16_t* vp = &Vt[t * 16 + bn_][bkg];
#pragma unroll
      for (int e = 0; e < 16; ++e) bv[e] = vp[e];
      o[t] = wmma_bf16(pa, bv, o[t]);
    }
    __syncthreads();  // before next tile overwrites Ks/Vt
  }

  // ---- epilogue: normalize, write ctx as [B,T,H,d] (== [B*T, C]) bf16 ----
  int crow = (lane >> 4) * 8, ccol = lane & 15;
#pragma unroll
  for (int r = 0; r < 8; ++r) {
    float inv = 1.0f / lrun[r];
    int trow = qt0 + crow + r;
    size_t base = (((size_t)b * SEQ + trow) * N_HEADS + h) * D_HEAD;
#pragma unroll
    for (int t = 0; t < 4; ++t)
      ctx[base + t * 16 + ccol] = (bf16_t)(o[t][r] * inv);
  }
}

// ---------------------------------------------------------------------------
// Output projection: ctx [8192,1024] x W_o [1024,1024] -> out f32
// ---------------------------------------------------------------------------
__global__ __launch_bounds__(256) void out_gemm(
    const bf16_t* __restrict__ X, const bf16_t* __restrict__ W,
    float* __restrict__ out) {
  __shared__ alignas(16) bf16_t As[64][40];
  __shared__ alignas(16) bf16_t Bst[128][40];
  int tid = threadIdx.x, lane = tid & 31, wave = tid >> 5;
  int wm = wave >> 2, wn = wave & 3;
  int bm = blockIdx.y * 64, bn = blockIdx.x * 128;
  v8f acc[2][2] = {};
  gemm_mainloop<D_MODEL>(X, W, bm, bn, tid, As, Bst, acc);

  int lrow = (lane >> 4) * 8, lcol = lane & 15;
#pragma unroll
  for (int i = 0; i < 2; ++i)
#pragma unroll
    for (int j = 0; j < 2; ++j)
#pragma unroll
      for (int r = 0; r < 8; ++r) {
        int m = bm + wm * 32 + i * 16 + lrow + r;
        int n = bn + wn * 32 + j * 16 + lcol;
        out[(size_t)m * D_MODEL + n] = acc[i][j][r];
      }
}

// ---------------------------------------------------------------------------
extern "C" void kernel_launch(void* const* d_in, const int* in_sizes, int n_in,
                              void* d_out, int out_size, void* d_ws,
                              size_t ws_size, hipStream_t stream) {
  (void)in_sizes; (void)n_in; (void)out_size; (void)ws_size;
  const float* x    = (const float*)d_in[0];
  const float* Wqkv = (const float*)d_in[1];
  const float* Wo   = (const float*)d_in[2];
  float* out = (float*)d_out;

  char* ws = (char*)d_ws;
  size_t off = 0;
  auto alloc = [&](size_t bytes) {
    void* p = ws + off;
    off += (bytes + 255) & ~(size_t)255;
    return p;
  };
  bf16_t* xb  = (bf16_t*)alloc((size_t)M_TOTAL * D_MODEL * 2);
  bf16_t* wqb = (bf16_t*)alloc((size_t)D_MODEL * N_QKV * 2);
  bf16_t* wob = (bf16_t*)alloc((size_t)D_MODEL * D_MODEL * 2);
  bf16_t* Qb  = (bf16_t*)alloc((size_t)M_TOTAL * D_MODEL * 2);
  bf16_t* Kb  = (bf16_t*)alloc((size_t)M_TOTAL * D_MODEL * 2);
  bf16_t* Vb  = (bf16_t*)alloc((size_t)M_TOTAL * D_MODEL * 2);
  bf16_t* ctx = (bf16_t*)alloc((size_t)M_TOTAL * D_MODEL * 2);

  cast_f32_bf16<<<1024, 256, 0, stream>>>(x, xb, M_TOTAL * D_MODEL);
  cast_f32_bf16<<<1024, 256, 0, stream>>>(Wqkv, wqb, D_MODEL * N_QKV);
  cast_f32_bf16<<<512, 256, 0, stream>>>(Wo, wob, D_MODEL * D_MODEL);

  qkv_gemm<<<dim3(N_QKV / 128, M_TOTAL / 64), 256, 0, stream>>>(
      xb, wqb, Qb, Kb, Vb);
  attn_kernel<<<dim3(SEQ / 128, BATCH * N_HEADS), 256, 0, stream>>>(
      Qb, Kb, Vb, ctx);
  out_gemm<<<dim3(D_MODEL / 128, M_TOTAL / 64), 256, 0, stream>>>(
      ctx, wob, out);
}
